// PolaLinearAttention_83803401880137
// MI455X (gfx1250) — matrix operand order repaired and checked by reading, not compiled, not run on hardware
//
#include <hip/hip_runtime.h>
#include <hip/hip_bf16.h>

// ---------------- types / helpers ----------------
typedef __attribute__((ext_vector_type(16))) __bf16 v16bf;
typedef __attribute__((ext_vector_type(8)))  float  v8f;

#define NTOK 16384   // N
#define CCH  384     // C
#define NHEAD 8
#define HDIM 48
#define BH   32      // B * HEADS
#define MROWS 65536  // B * N

__device__ __forceinline__ __bf16 f2bf(float f) { return (__bf16)f; }

// pack two floats -> two bf16 -> one 32-bit store (v_cvt_pk_bf16_f32 + b32 store)
__device__ __forceinline__ void store_bf16x2(__bf16* p, float a, float b) {
    unsigned lo = (unsigned)__builtin_bit_cast(unsigned short, (__bf16)a);
    unsigned hi = (unsigned)__builtin_bit_cast(unsigned short, (__bf16)b);
    *reinterpret_cast<unsigned*>(p) = (hi << 16) | lo;
}

__device__ __forceinline__ float softplusf_(float x) {
    return (x > 20.f) ? x : log1pf(__expf(x));
}
__device__ __forceinline__ float sigmoidf_(float x) {
    return 1.f / (1.f + __expf(-x));
}
// q_pos/q_neg in one pow: only one side of relu_pow(x), relu_pow(-x) is nonzero.
__device__ __forceinline__ void polar_pow(float x, float p, float& pos, float& neg) {
    float a  = fabsf(x);
    float pw = (a > 0.f) ? __powf(a, p) : 0.f;
    pos = (x > 0.f) ? pw : 0.f;
    neg = (x < 0.f) ? pw : 0.f;
}

// A fragment: 16x32 (MxK) bf16, source row-major s[m*ld + k].
// lanes 0-15: M=lane, K={0..7,16..23}; lanes 16-31: M=lane-16, K={8..15,24..31}.
__device__ __forceinline__ v16bf load_frag_a(const __bf16* s, int ld) {
    int lane = threadIdx.x & 31;
    int m  = lane & 15;
    int kb = (lane >> 4) << 3;
    const __bf16* r0 = s + m * ld + kb;
    v16bf a;
#pragma unroll
    for (int e = 0; e < 8; ++e) { a[e] = r0[e]; a[e + 8] = r0[e + 16]; }
    return a;
}

// B fragment: 32x16 (KxN) bf16, source row-major s[k*ld + n].
// lanes 0-15: N=lane, K={0..7,16..23}; lanes 16-31: N=lane-16, K={8..15,24..31}.
__device__ __forceinline__ v16bf load_frag_b(const __bf16* s, int ld) {
    int lane = threadIdx.x & 31;
    int n  = lane & 15;
    int kb = (lane >> 4) << 3;
    const __bf16* r = s + kb * ld + n;
    v16bf b;
#pragma unroll
    for (int e = 0; e < 8; ++e) { b[e] = r[e * ld]; b[e + 8] = r[(e + 16) * ld]; }
    return b;
}

__device__ __forceinline__ v8f wmma_bf16(v16bf a, v16bf b, v8f c) {
    return __builtin_amdgcn_wmma_f32_16x16x32_bf16(
        false, a, false, b, (short)0, c, false, false);
}

// ---------------- kernel 0: zero scratch accumulators ----------------
__global__ void k_zero(float* __restrict__ p, int n) {
    int i = blockIdx.x * 256 + threadIdx.x;
    if (i < n) p[i] = 0.f;
}

// ---------------- kernel 1: fused QG/KV GEMM + polarity epilogue ----------------
// y = x @ [W_qg | W_kv]  (M=65536, K=384, Ncols=1536). 128x32 block tile,
// 8 waves: wave = M tile, 2 N tiles per wave (2 WMMA per staged B tile).
__global__ __launch_bounds__(256) void k_qkv_gemm(
    const float* __restrict__ x, const float* __restrict__ Wqg,
    const float* __restrict__ Wkv, const float* __restrict__ pos_enc,
    const float* __restrict__ power_p, const float* __restrict__ scale_p,
    __bf16* __restrict__ QSIM, __bf16* __restrict__ KK,
    __bf16* __restrict__ VB, __bf16* __restrict__ G)
{
    const int LDA = 40, LDB = 40;
    __shared__ __bf16 Al[128 * 40];
    __shared__ __bf16 Bl[32 * 40];
    int tid = threadIdx.x;
    int m0 = blockIdx.x * 128;
    int n0 = blockIdx.y * 32;
    int w = tid >> 5;                 // M tile of this wave
    v8f acc0 = {}, acc1 = {};
    for (int k0 = 0; k0 < CCH; k0 += 32) {
        __syncthreads();
#pragma unroll
        for (int i = 0; i < 8; ++i) {               // A: 128x32, packed pairs
            int idx = tid + i * 256;
            int r = idx >> 4, kp = idx & 15;
            float2 xv = *reinterpret_cast<const float2*>(
                &x[(size_t)(m0 + r) * CCH + k0 + 2 * kp]);
            store_bf16x2(&Al[r * LDA + 2 * kp], xv.x, xv.y);
        }
#pragma unroll
        for (int i = 0; i < 2; ++i) {               // B: 32x32, packed pairs
            int idx = tid + i * 256;
            int r = idx >> 4, cp = idx & 15;
            int c = 2 * cp, col = n0 + c;
            const float* wsrc = (col < 768)
                ? &Wqg[(size_t)(k0 + r) * 768 + col]
                : &Wkv[(size_t)(k0 + r) * 768 + col - 768];
            float2 wv = *reinterpret_cast<const float2*>(wsrc);
            store_bf16x2(&Bl[r * LDB + c], wv.x, wv.y);
        }
        if (k0 + 32 < CCH)                          // speculative next-slice prefetch
            __builtin_prefetch(&x[(size_t)(m0 + (tid & 127)) * CCH + k0 + 32], 0, 1);
        __syncthreads();
        v16bf a = load_frag_a(&Al[w * 16 * LDA], LDA);
        acc0 = wmma_bf16(a, load_frag_b(&Bl[0],  LDB), acc0);
        acc1 = wmma_bf16(a, load_frag_b(&Bl[16], LDB), acc1);
    }
    // epilogue: per-lane column fixed per N tile, rows vary with v
    int lane = tid & 31;
    int rbase = m0 + w * 16 + ((lane >> 4) << 3);
#pragma unroll
    for (int nt = 0; nt < 2; ++nt) {
        const v8f acc = nt ? acc1 : acc0;
        int col = n0 + nt * 16 + (lane & 15);
        if (col < 384) {                              // q channel
            int c = col, hh = c / HDIM, d = c - hh * HDIM;
            float inv_sc = 1.f / softplusf_(scale_p[c]);
            float p = 1.f + 4.f * sigmoidf_(power_p[hh * HDIM + d]);
#pragma unroll
            for (int v = 0; v < 8; ++v) {
                int m = rbase + v;
                int b_ = m >> 14, tok = m & (NTOK - 1);
                float qpos, qneg;
                polar_pow(acc[v] * inv_sc, p, qpos, qneg);
                size_t o = ((size_t)(b_ * NHEAD + hh) * NTOK + tok) * 96;
                QSIM[o + d]        = f2bf(qpos);
                QSIM[o + HDIM + d] = f2bf(qneg);
            }
        } else if (col < 768) {                       // gate
            int c = col - 384;
#pragma unroll
            for (int v = 0; v < 8; ++v) {
                int m = rbase + v;
                G[(size_t)m * CCH + c] = f2bf(acc[v]);
            }
        } else if (col < 1152) {                      // k channel
            int c = col - 768, hh = c / HDIM, d = c - hh * HDIM;
            float inv_sc = 1.f / softplusf_(scale_p[c]);
            float p = 1.f + 4.f * sigmoidf_(power_p[hh * HDIM + d]);
#pragma unroll
            for (int v = 0; v < 8; ++v) {
                int m = rbase + v;
                int b_ = m >> 14, tok = m & (NTOK - 1);
                float kpos, kneg;
                polar_pow((acc[v] + pos_enc[(size_t)tok * CCH + c]) * inv_sc,
                          p, kpos, kneg);
                size_t o = ((size_t)(b_ * NHEAD + hh) * NTOK + tok) * 96;
                KK[o + d]        = f2bf(kpos);
                KK[o + HDIM + d] = f2bf(kneg);
            }
        } else {                                      // v channel
            int c = col - 1152, hh = c / HDIM, d = c - hh * HDIM;
#pragma unroll
            for (int v = 0; v < 8; ++v) {
                int m = rbase + v;
                int b_ = m >> 14, tok = m & (NTOK - 1);
                VB[((size_t)(b_ * NHEAD + hh) * NTOK + tok) * HDIM + d] = f2bf(acc[v]);
            }
        }
    }
}

// ---------------- kernel 2: KV = kk^T @ v (96x48 per head) + k_sum ----------------
// 192 threads = 6 waves, every wave computes (no divergent WMMA, EXEC all-ones).
// kk staged TRANSPOSED into LDS so A fragments read contiguous runs.
// k_sum: 192 % 48 == 0 -> each thread owns a fixed kk column pair in registers.
__global__ __launch_bounds__(192) void k_kv_reduce(
    const __bf16* __restrict__ KK, const __bf16* __restrict__ VB,
    float* __restrict__ KV, float* __restrict__ KSUM)
{
    const int LDT = 40, LDV = 56;
    __shared__ __bf16 Klt[96 * 40];    // kk^T tile: [d][n_local]
    __shared__ __bf16 Vl[32 * 56];
    int tid = threadIdx.x;
    int bh = blockIdx.y;
    int nbase = blockIdx.x * 1024;
    int w = tid >> 5;                  // 0..5, all compute
    int dpc = tid % 48;                // fixed kk column pair for k_sum
    float s0 = 0.f, s1 = 0.f;
    v8f acc0 = {}, acc1 = {}, acc2 = {};
    for (int kk0 = 0; kk0 < 1024; kk0 += 32) {
        __syncthreads();
#pragma unroll
        for (int i = 0; i < 8; ++i) {              // kk: 32 rows x 48 u32 pairs
            int idx = tid + i * 192;
            int r = idx / 48, dp = idx % 48;       // dp == dpc for every i
            unsigned u = *reinterpret_cast<const unsigned*>(
                KK + ((size_t)bh * NTOK + nbase + kk0 + r) * 96 + 2 * dp);
            Klt[(2 * dp)     * LDT + r] = __builtin_bit_cast(__bf16, (unsigned short)(u & 0xffffu));
            Klt[(2 * dp + 1) * LDT + r] = __builtin_bit_cast(__bf16, (unsigned short)(u >> 16));
            s0 += __builtin_bit_cast(float, u << 16);          // bf16 -> f32 by shift
            s1 += __builtin_bit_cast(float, u & 0xffff0000u);
        }
#pragma unroll
        for (int i = 0; i < 4; ++i) {              // v: 32 rows x 24 u32 pairs
            int idx = tid + i * 192;
            int r = idx / 24, ep = idx % 24;
            *reinterpret_cast<unsigned*>(&Vl[r * LDV + 2 * ep]) =
                *reinterpret_cast<const unsigned*>(
                    VB + ((size_t)bh * NTOK + nbase + kk0 + r) * HDIM + 2 * ep);
        }
        __syncthreads();
        v16bf a = load_frag_a(&Klt[(w * 16) * LDT], LDT);   // contiguous runs
        acc0 = wmma_bf16(a, load_frag_b(&Vl[0],  LDV), acc0);
        acc1 = wmma_bf16(a, load_frag_b(&Vl[16], LDV), acc1);
        acc2 = wmma_bf16(a, load_frag_b(&Vl[32], LDV), acc2);
    }
    int lane = tid & 31;
    int nc = lane & 15;
    int rt = (lane >> 4) << 3;
#pragma unroll
    for (int v = 0; v < 8; ++v) {
        int d = w * 16 + rt + v;
        float* kvrow = &KV[((size_t)bh * 96 + d) * 48];
        atomicAdd(kvrow + 0  + nc, acc0[v]);
        atomicAdd(kvrow + 16 + nc, acc1[v]);
        atomicAdd(kvrow + 32 + nc, acc2[v]);
    }
    atomicAdd(&KSUM[bh * 96 + 2 * dpc],     s0);
    atomicAdd(&KSUM[bh * 96 + 2 * dpc + 1], s1);
}

// ---------------- kernel 3: x_out = q_sim @ B' with per-row z scaling ----------------
// B'[d][e] = KV[e<24 ? d : (d+48)%96][e] * inv_n  (folds q_opp via half-swap).
__global__ __launch_bounds__(256) void k_attn(
    const __bf16* __restrict__ QSIM, const float* __restrict__ KV,
    const float* __restrict__ KSUM, __bf16* __restrict__ ATT)
{
    const int LDBM = 56;
    __shared__ __bf16 Bm[96 * 56];
    __shared__ float  kmean[96], kmeanSw[96];
    __shared__ float  zbuf[8][16][2];
    int tid = threadIdx.x;
    int bh = blockIdx.x;
    int b_ = bh >> 3, h_ = bh & 7;
    const float inv_n = 1.0f / (float)NTOK;
    for (int idx = tid; idx < 96 * 24; idx += 256) {   // packed pair stores
        int d = idx / 24, ep = idx - d * 24;
        int e = 2 * ep;
        int dd = (e < 24) ? d : ((d + 48) % 96);       // same dd for e, e+1
        const float* kvp = &KV[((size_t)bh * 96 + dd) * 48 + e];
        store_bf16x2(&Bm[d * LDBM + e], kvp[0] * inv_n, kvp[1] * inv_n);
    }
    if (tid < 96) {
        kmean[tid]   = KSUM[bh * 96 + tid] * inv_n;
        kmeanSw[tid] = KSUM[bh * 96 + ((tid + 48) % 96)] * inv_n;
    }
    __syncthreads();
    int w = tid >> 5, lane = tid & 31;
    int t0 = (blockIdx.y * 8 + w) * 16;
    const __bf16* qbase = QSIM + ((size_t)bh * NTOK + t0) * 96;
    if (lane < 16) {                                   // per-token z dots
        const __bf16* qr = qbase + (size_t)lane * 96;
        float s1 = 0.f, s2 = 0.f;
#pragma unroll 4
        for (int d = 0; d < 96; ++d) {
            float q = (float)qr[d];
            s1 += q * kmean[d];
            s2 += q * kmeanSw[d];
        }
        zbuf[w][lane][0] = 1.f / (s1 + 1e-6f);
        zbuf[w][lane][1] = 1.f / (s2 + 1e-6f);
    }
    __syncthreads();
    v8f acc0 = {}, acc1 = {}, acc2 = {};
#pragma unroll
    for (int k0 = 0; k0 < 96; k0 += 32) {
        v16bf a = load_frag_a(qbase + k0, 96);         // A straight from global
        acc0 = wmma_bf16(a, load_frag_b(&Bm[k0 * LDBM + 0],  LDBM), acc0);
        acc1 = wmma_bf16(a, load_frag_b(&Bm[k0 * LDBM + 16], LDBM), acc1);
        acc2 = wmma_bf16(a, load_frag_b(&Bm[k0 * LDBM + 32], LDBM), acc2);
    }
    int nc = lane & 15, rt = (lane >> 4) << 3;
#pragma unroll
    for (int v = 0; v < 8; ++v) {
        int m = rt + v;
        int tok = t0 + m;
        float zs = zbuf[w][m][0], zo = zbuf[w][m][1];
        size_t row = ((size_t)b_ * NTOK + tok) * CCH + h_ * HDIM;
        ATT[row + nc]      = f2bf(acc0[v] * zs);
        ATT[row + 16 + nc] = f2bf(acc1[v] * ((nc + 16 < 24) ? zs : zo));
        ATT[row + 32 + nc] = f2bf(acc2[v] * zo);
    }
}

// ---------------- kernel 4: depthwise 5x5 conv + gate fuse ----------------
__global__ __launch_bounds__(256) void k_conv_gate(
    const __bf16* __restrict__ VB, const float* __restrict__ dwc_w,
    const float* __restrict__ dwc_b, const __bf16* __restrict__ ATT,
    const __bf16* __restrict__ G, __bf16* __restrict__ PRE)
{
    int d  = blockIdx.y;
    int bh = blockIdx.z;
    int b_ = bh >> 3, h_ = bh & 7;
    int pid = blockIdx.x * 256 + threadIdx.x;
    int y = pid >> 7, xp = pid & 127;
    float wloc[25];
#pragma unroll
    for (int i = 0; i < 25; ++i) wloc[i] = dwc_w[d * 25 + i];
    float s = dwc_b[d];
    const __bf16* vb = VB + (size_t)bh * NTOK * HDIM + d;
#pragma unroll
    for (int ky = 0; ky < 5; ++ky) {
        int yy = y + ky - 2;
        if (yy < 0 || yy > 127) continue;
#pragma unroll
        for (int kx = 0; kx < 5; ++kx) {
            int xx = xp + kx - 2;
            if (xx < 0 || xx > 127) continue;
            s += (float)vb[(size_t)(yy * 128 + xx) * HDIM] * wloc[ky * 5 + kx];
        }
    }
    size_t mi = ((size_t)b_ * NTOK + pid) * CCH + h_ * HDIM + d;
    PRE[mi] = f2bf(((float)ATT[mi] + s) * (float)G[mi]);
}

// ---------------- kernel 5: final projection GEMM + bias ----------------
__global__ __launch_bounds__(256) void k_proj_gemm(
    const __bf16* __restrict__ PRE, const float* __restrict__ Wp,
    const float* __restrict__ bp, float* __restrict__ out)
{
    const int LDA = 40, LDB = 40;
    __shared__ __bf16 Al[128 * 40];
    __shared__ __bf16 Bl[32 * 40];
    int tid = threadIdx.x;
    int m0 = blockIdx.x * 128;
    int n0 = blockIdx.y * 32;
    int w = tid >> 5;
    v8f acc0 = {}, acc1 = {};
    for (int k0 = 0; k0 < CCH; k0 += 32) {
        __syncthreads();
#pragma unroll
        for (int i = 0; i < 8; ++i) {               // A: bf16 -> u32 copy
            int idx = tid + i * 256;
            int r = idx >> 4, kp = idx & 15;
            *reinterpret_cast<unsigned*>(&Al[r * LDA + 2 * kp]) =
                *reinterpret_cast<const unsigned*>(
                    &PRE[(size_t)(m0 + r) * CCH + k0 + 2 * kp]);
        }
#pragma unroll
        for (int i = 0; i < 2; ++i) {               // B: packed f32 pairs
            int idx = tid + i * 256;
            int r = idx >> 4, cp = idx & 15;
            int c = 2 * cp;
            float2 wv = *reinterpret_cast<const float2*>(
                &Wp[(size_t)(k0 + r) * CCH + n0 + c]);
            store_bf16x2(&Bl[r * LDB + c], wv.x, wv.y);
        }
        if (k0 + 32 < CCH)
            __builtin_prefetch(&PRE[(size_t)(m0 + (tid & 127)) * CCH + k0 + 32], 0, 1);
        __syncthreads();
        v16bf a = load_frag_a(&Al[w * 16 * LDA], LDA);
        acc0 = wmma_bf16(a, load_frag_b(&Bl[0],  LDB), acc0);
        acc1 = wmma_bf16(a, load_frag_b(&Bl[16], LDB), acc1);
    }
    int lane = tid & 31;
    int rbase = m0 + w * 16 + ((lane >> 4) << 3);
#pragma unroll
    for (int nt = 0; nt < 2; ++nt) {
        const v8f acc = nt ? acc1 : acc0;
        int col = n0 + nt * 16 + (lane & 15);
        float bias = bp[col];
#pragma unroll
        for (int v = 0; v < 8; ++v)
            out[(size_t)(rbase + v) * CCH + col] = acc[v] + bias;
    }
}

// ---------------- launcher ----------------
extern "C" void kernel_launch(void* const* d_in, const int* in_sizes, int n_in,
                              void* d_out, int out_size, void* d_ws, size_t ws_size,
                              hipStream_t stream) {
    (void)in_sizes; (void)n_in; (void)out_size; (void)ws_size;
    const float* x     = (const float*)d_in[0];
    const float* Wqg   = (const float*)d_in[1];
    const float* Wkv   = (const float*)d_in[2];
    const float* Wproj = (const float*)d_in[3];
    const float* bproj = (const float*)d_in[4];
    const float* pos   = (const float*)d_in[5];
    const float* powp  = (const float*)d_in[6];
    const float* sclp  = (const float*)d_in[7];
    const float* dwcw  = (const float*)d_in[8];
    const float* dwcb  = (const float*)d_in[9];
    float* out = (float*)d_out;

    char* ws = (char*)d_ws;
    size_t off = 0;
    auto alloc = [&](size_t bytes) -> char* {
        char* p = ws + off;
        off += (bytes + 255) & ~(size_t)255;
        return p;
    };
    __bf16* QSIM = (__bf16*)alloc((size_t)BH * NTOK * 96 * 2);
    __bf16* KKb  = (__bf16*)alloc((size_t)BH * NTOK * 96 * 2);
    __bf16* VBb  = (__bf16*)alloc((size_t)BH * NTOK * HDIM * 2);
    __bf16* Gb   = (__bf16*)alloc((size_t)MROWS * CCH * 2);
    __bf16* ATTb = (__bf16*)alloc((size_t)MROWS * CCH * 2);
    __bf16* PREb = (__bf16*)alloc((size_t)MROWS * CCH * 2);
    float*  KV   = (float*)alloc((size_t)BH * 96 * 48 * 4);
    float*  KSUM = (float*)alloc((size_t)BH * 96 * 4);

    k_zero<<<dim3((BH * 96 * 48 + 255) / 256), dim3(256), 0, stream>>>(KV, BH * 96 * 48);
    k_zero<<<dim3((BH * 96 + 255) / 256),      dim3(256), 0, stream>>>(KSUM, BH * 96);

    k_qkv_gemm<<<dim3(MROWS / 128, 1536 / 32), dim3(256), 0, stream>>>(
        x, Wqg, Wkv, pos, powp, sclp, QSIM, KKb, VBb, Gb);
    k_kv_reduce<<<dim3(NTOK / 1024, BH), dim3(192), 0, stream>>>(KKb, VBb, KV, KSUM);
    k_attn<<<dim3(BH, NTOK / 128), dim3(256), 0, stream>>>(QSIM, KV, KSUM, ATTb);
    k_conv_gate<<<dim3(NTOK / 256, HDIM, BH), dim3(256), 0, stream>>>(
        VBb, dwcw, dwcb, ATTb, Gb, PREb);
    k_proj_gemm<<<dim3(MROWS / 128, CCH / 32), dim3(256), 0, stream>>>(
        PREb, Wproj, bproj, out);
}
